// Head_46377056862332
// MI455X (gfx1250) — compile-verified, matrix operand order checked by reading, not run on hardware
//
#include <hip/hip_runtime.h>
#include <hip/hip_bf16.h>
#include <cstdint>
#include <cstddef>

// CDNA5 / gfx1250 WMMA fragment types (wave32).
typedef __attribute__((ext_vector_type(16))) _Float16 v16h;
typedef __attribute__((ext_vector_type(8)))  float    v8f;
typedef __attribute__((ext_vector_type(4)))  int      v4i;

static constexpr int T     = 256;   // sequence length
static constexpr int CEMB  = 384;   // embed dim
static constexpr int HS    = 64;    // head size
static constexpr int NQKV  = 192;   // 3 * HS

#if __has_builtin(__builtin_amdgcn_global_load_async_to_lds_b128)
#define HAVE_ASYNC_LDS 1
// From the compiler diagnostic: the builtin takes a pointer to int4 in the
// global (AS1) space and a pointer to int4 in LDS (AS3), plus imm offset/cpol.
typedef __attribute__((address_space(1))) v4i* gv4i_p;   // int4 in global memory
typedef __attribute__((address_space(3))) v4i* lv4i_p;   // int4 in LDS
#endif

// ISA 7.12.2: 16-bit A (16x32) / B (32x16) fragment: lane L holds M/N = L%16;
// element e maps to K index: e=0..7 -> K=e (+8 if lane>=16), e=8..15 -> K=e+8 (+8 if lane>=16).
__device__ __forceinline__ int kmap(int e, int hi) {
    return (e & 7) + ((e >> 3) << 4) + (hi ? 8 : 0);
}

// ---------------------------------------------------------------------------
// Kernel 1: QKV projection. [B*T,384] f32 @ [384,192] f32 -> Q,K,V f16.
// Q is pre-scaled by (1/sqrt(384)) * log2(e) so attention softmax runs in
// exp2 domain (native v_exp_f32).
// ---------------------------------------------------------------------------
__global__ __launch_bounds__(256) void qkv_proj_kernel(
    const float* __restrict__ x, const float* __restrict__ w,
    _Float16* __restrict__ qb, _Float16* __restrict__ kb, _Float16* __restrict__ vb)
{
    constexpr int LDP = 72;                 // 144B row stride: 16B-multiple, conflict-free
    __shared__ _Float16 xs[128 * LDP];      // x tile: 128 rows x 32 K
    __shared__ _Float16 wt[NQKV * LDP];     // w chunk, transposed: 192 cols x 32 K

    const int tid = threadIdx.x;
    const int wv  = tid >> 5;               // wave 0..7
    const int ln  = tid & 15;
    const int hi  = (tid >> 4) & 1;         // lane >= 16 within wave
    const size_t m0 = (size_t)blockIdx.x * 128;

    v8f acc[12] = {};                       // wave strip: 16 rows x 192 cols

    const int xr = tid & 127;               // x loader: row
    const int xc = (tid >> 7) * 16;         // x loader: col base (16 floats)
    const int wr = tid >> 3;                // w loader: K row 0..31
    const int wc = (tid & 7) * 24;          // w loader: col base (24 floats)

    for (int kk = 0; kk < 12; ++kk) {
        const int k0 = kk * 32;

        // Prefetch next x chunk while this one converts (global_prefetch_b8).
        if (kk < 11)
            __builtin_prefetch((const void*)(x + (m0 + xr) * CEMB + (k0 + 32) + xc), 0, 3);

        // x[128 x 32] f32 -> f16 LDS
        const float4* xg = reinterpret_cast<const float4*>(x + (m0 + xr) * CEMB + k0 + xc);
#pragma unroll
        for (int p = 0; p < 4; ++p) {
            float4 f = xg[p];
            xs[xr * LDP + xc + 4 * p + 0] = (_Float16)f.x;
            xs[xr * LDP + xc + 4 * p + 1] = (_Float16)f.y;
            xs[xr * LDP + xc + 4 * p + 2] = (_Float16)f.z;
            xs[xr * LDP + xc + 4 * p + 3] = (_Float16)f.w;
        }
        // w[32 x 192] f32 -> transposed f16 LDS (B fragments read contiguous K runs)
        const float4* wg = reinterpret_cast<const float4*>(w + (size_t)(k0 + wr) * NQKV + wc);
#pragma unroll
        for (int p = 0; p < 6; ++p) {
            float4 f = wg[p];
            wt[(wc + 4 * p + 0) * LDP + wr] = (_Float16)f.x;
            wt[(wc + 4 * p + 1) * LDP + wr] = (_Float16)f.y;
            wt[(wc + 4 * p + 2) * LDP + wr] = (_Float16)f.z;
            wt[(wc + 4 * p + 3) * LDP + wr] = (_Float16)f.w;
        }
        __syncthreads();

        v16h af;
#pragma unroll
        for (int e = 0; e < 16; ++e) af[e] = xs[(wv * 16 + ln) * LDP + kmap(e, hi)];
#pragma unroll
        for (int j = 0; j < 12; ++j) {
            v16h bf;
#pragma unroll
            for (int e = 0; e < 16; ++e) bf[e] = wt[(j * 16 + ln) * LDP + kmap(e, hi)];
            acc[j] = __builtin_amdgcn_wmma_f32_16x16x32_f16(
                false, af, false, bf, (short)0, acc[j], false, false);
        }
        __syncthreads();
    }

    // qscale = log2(e) / sqrt(C): fold softmax scale + exp2 conversion into Q.
    const float qscale = 1.442695040888963f * rsqrtf((float)CEMB);
#pragma unroll
    for (int j = 0; j < 12; ++j) {
#pragma unroll
        for (int i = 0; i < 8; ++i) {
            const size_t gm = m0 + wv * 16 + i + 8 * hi;   // C layout: row = i + 8*hi
            const int    c  = j * 16 + ln;                 // col = lane%16
            const float  v  = acc[j][i];
            if (j < 4)      qb[gm * HS + c]         = (_Float16)(v * qscale);
            else if (j < 8) kb[gm * HS + (c - 64)]  = (_Float16)v;
            else            vb[gm * HS + (c - 128)] = (_Float16)v;
        }
    }
}

// ---------------------------------------------------------------------------
// Kernel 2: causal flash attention, one workgroup per batch element.
// 8 waves; wave w owns the 16-row strip PAIR {w, 15-w}: strip a needs
// floor(a/2)+1 causal 32-col chunks, so every wave does exactly 9 chunks
// (balanced). No workgroup barriers inside the divergent chunk loops.
// Q/K staged to LDS with CDNA5 async global->LDS DMA (ASYNCcnt); V is
// transposed through VGPRs while the DMA runs.
// ---------------------------------------------------------------------------
__global__ __launch_bounds__(256) void attn_kernel(
    const _Float16* __restrict__ qb, const _Float16* __restrict__ kb,
    const _Float16* __restrict__ vb, float* __restrict__ out)
{
    constexpr int KLDP = 72;    // Q/K rows [256][72]  (144B stride, 16B-mult, conflict-free)
    constexpr int VLDP = 264;   // V^T rows [64][264]  (528B stride, 16B-mult)
    constexpr int PLDP = 40;    // P stage  [8][16][40]
    extern __shared__ _Float16 smem[];
    _Float16* Qs = smem;                    // 256*72 = 18432 halves
    _Float16* Ks = Qs + 256 * KLDP;         // 256*72 = 18432 halves
    _Float16* Vt = Ks + 256 * KLDP;         // 64*264 = 16896 halves
    _Float16* Ps = Vt + 64 * VLDP;          // 8*16*40 = 5120 halves

    const int b   = blockIdx.x;
    const int tid = threadIdx.x;
    const int wv  = tid >> 5;
    const int ln  = tid & 15;
    const int hi  = (tid >> 4) & 1;

    // ---- Stage Q, K (row-major) and V (transposed) into LDS, one row/thread.
    {
        const int r = tid; // 0..255
        const _Float16* qg = qb + ((size_t)b * T + r) * HS;
        const _Float16* kg = kb + ((size_t)b * T + r) * HS;
        const float4*   vg = reinterpret_cast<const float4*>(vb + ((size_t)b * T + r) * HS);

#ifdef HAVE_ASYNC_LDS
        // 64 halves = 128B per row = 8 async B128 transfers; DMA writes LDS
        // directly (no VGPR round trip), overlapping the V transpose below.
#pragma unroll
        for (int j = 0; j < 8; ++j) {
            __builtin_amdgcn_global_load_async_to_lds_b128(
                (gv4i_p)(qg + j * 8), (lv4i_p)(&Qs[r * KLDP + j * 8]), 0, 0);
            __builtin_amdgcn_global_load_async_to_lds_b128(
                (gv4i_p)(kg + j * 8), (lv4i_p)(&Ks[r * KLDP + j * 8]), 0, 0);
        }
#else
        const float4* qg4 = reinterpret_cast<const float4*>(qg);
        const float4* kg4 = reinterpret_cast<const float4*>(kg);
#pragma unroll
        for (int j = 0; j < 8; ++j) {
            *reinterpret_cast<float4*>(&Qs[r * KLDP + j * 8]) = qg4[j];
            *reinterpret_cast<float4*>(&Ks[r * KLDP + j * 8]) = kg4[j];
        }
#endif
        // V transpose (element scatter) must pass through VGPRs.
#pragma unroll
        for (int j = 0; j < 8; ++j) {
            float4 vf = vg[j];
            const _Float16* hp = reinterpret_cast<const _Float16*>(&vf);
#pragma unroll
            for (int i2 = 0; i2 < 8; ++i2) Vt[(j * 8 + i2) * VLDP + r] = hp[i2];
        }
    }
#ifdef HAVE_ASYNC_LDS
    asm volatile("s_wait_asynccnt 0" ::: "memory");
#endif
    __syncthreads();

    // ---- Balanced strip pair: pi=0 -> strip wv, pi=1 -> strip 15-wv.
#pragma unroll
    for (int pi = 0; pi < 2; ++pi) {
        const int a  = pi ? (15 - wv) : wv;   // strip index, rows [16a, 16a+16)
        const int t0 = a * 16;

        // Q strip as A fragments: 2 K-chunks of 32 (contiguous LDS runs).
        v16h qa[2];
#pragma unroll
        for (int kc = 0; kc < 2; ++kc)
#pragma unroll
            for (int e = 0; e < 16; ++e)
                qa[kc][e] = Qs[(t0 + ln) * KLDP + 32 * kc + kmap(e, hi)];

        v8f   o[4] = {};                      // 16 rows x 64 cols
        float mrow[8], lrow[8];
#pragma unroll
        for (int i = 0; i < 8; ++i) { mrow[i] = -3.0e38f; lrow[i] = 0.0f; }

        const int nch = (a >> 1) + 1;         // causal 32-col chunks for this strip
        for (int ch = 0; ch < nch; ++ch) {
            const int s0 = ch * 32;

            // S = Q (16x64) @ K^T (64x32): 2 frags x 2 K-steps = 4 WMMA
            v8f sf[2] = {};
#pragma unroll
            for (int ct = 0; ct < 2; ++ct)
#pragma unroll
                for (int kc = 0; kc < 2; ++kc) {
                    v16h kf;
#pragma unroll
                    for (int e = 0; e < 16; ++e)
                        kf[e] = Ks[(s0 + 16 * ct + ln) * KLDP + 32 * kc + kmap(e, hi)];
                    sf[ct] = __builtin_amdgcn_wmma_f32_16x16x32_f16(
                        false, qa[kc], false, kf, (short)0, sf[ct], false, false);
                }

            // Causal mask + online softmax (exp2 domain; scale folded into Q).
#pragma unroll
            for (int i = 0; i < 8; ++i) {
                const int trow = t0 + i + 8 * hi;
#pragma unroll
                for (int ct = 0; ct < 2; ++ct) {
                    const int scol = s0 + 16 * ct + ln;
                    if (scol > trow) sf[ct][i] = -3.0e38f;
                }
                float rm = fmaxf(sf[0][i], sf[1][i]);
#pragma unroll
                for (int off = 1; off < 16; off <<= 1) rm = fmaxf(rm, __shfl_xor(rm, off, 32));
                const float mold = mrow[i];
                const float mnew = fmaxf(mold, rm);
                const float corr = exp2f(mold - mnew);
                mrow[i] = mnew;
                float rs = 0.0f;
#pragma unroll
                for (int ct = 0; ct < 2; ++ct) {
                    const float p = exp2f(sf[ct][i] - mnew);
                    sf[ct][i] = p;
                    rs += p;
                }
#pragma unroll
                for (int off = 1; off < 16; off <<= 1) rs += __shfl_xor(rs, off, 32);
                lrow[i] = lrow[i] * corr + rs;
#pragma unroll
                for (int dc = 0; dc < 4; ++dc) o[dc][i] *= corr;
            }

            // C-layout -> A-layout via per-wave LDS staging; LDS is in-order
            // per wave, explicit CDNA5 split-counter wait for the RAW.
#pragma unroll
            for (int ct = 0; ct < 2; ++ct)
#pragma unroll
                for (int i = 0; i < 8; ++i)
                    Ps[(wv * 16 + i + 8 * hi) * PLDP + 16 * ct + ln] = (_Float16)sf[ct][i];

            asm volatile("s_wait_dscnt 0" ::: "memory");

            v16h pa;
#pragma unroll
            for (int e = 0; e < 16; ++e)
                pa[e] = Ps[(wv * 16 + ln) * PLDP + kmap(e, hi)];

            // O += P (16x32) @ V (32x64): 4 WMMA
#pragma unroll
            for (int dc = 0; dc < 4; ++dc) {
                v16h vf;
#pragma unroll
                for (int e = 0; e < 16; ++e)
                    vf[e] = Vt[(16 * dc + ln) * VLDP + s0 + kmap(e, hi)];
                o[dc] = __builtin_amdgcn_wmma_f32_16x16x32_f16(
                    false, pa, false, vf, (short)0, o[dc], false, false);
            }
        }

        // Normalize and store f32 output rows of this strip.
#pragma unroll
        for (int i = 0; i < 8; ++i) {
            const float inv  = 1.0f / lrow[i];
            const int   trow = t0 + i + 8 * hi;
            float* orow = out + ((size_t)b * T + trow) * HS;
#pragma unroll
            for (int dc = 0; dc < 4; ++dc) orow[16 * dc + ln] = o[dc][i] * inv;
        }
    }
}

// ---------------------------------------------------------------------------
extern "C" void kernel_launch(void* const* d_in, const int* in_sizes, int n_in,
                              void* d_out, int out_size, void* d_ws, size_t ws_size,
                              hipStream_t stream) {
    const float* x = (const float*)d_in[0];
    const float* w = (const float*)d_in[1];
    float* out = (float*)d_out;

    const int B = in_sizes[0] / (T * CEMB);   // 512

    // Workspace: Q, K, V as f16, each B*T*HS elements (~16.8 MB each).
    _Float16* qb = (_Float16*)d_ws;
    _Float16* kb = qb + (size_t)B * T * HS;
    _Float16* vb = kb + (size_t)B * T * HS;

    qkv_proj_kernel<<<dim3((unsigned)((size_t)B * T / 128)), dim3(256), 0, stream>>>(
        x, w, qb, kb, vb);

    const size_t smem =
        (size_t)(256 * 72 + 256 * 72 + 64 * 264 + 8 * 16 * 40) * sizeof(_Float16); // 117760 B
    (void)hipFuncSetAttribute((const void*)attn_kernel,
                              hipFuncAttributeMaxDynamicSharedMemorySize, (int)smem);
    attn_kernel<<<dim3((unsigned)B), dim3(256), smem, stream>>>(qb, kb, vb, out);
}
